// CPN_16166256902279
// MI455X (gfx1250) — compile-verified, matrix operand order checked by reading, not run on hardware
//
#include <hip/hip_runtime.h>
#include <hip/hip_bf16.h>

// ---------------- CDNA5 WMMA types ----------------
typedef __attribute__((ext_vector_type(16))) __bf16 v16bf;
typedef __attribute__((ext_vector_type(8)))  float  v8f;

#define HC 256
#define WC 256
#define HWC 65536      // 256*256
#define HF 512
#define WF 512
#define HWF 262144     // 512*512
#define TOPK 512
#define SAMPLES 64
#define ORDER 6

// K-index map for 16-bit A (16x32) and B (32x16) fragments (wave32).
// lane half h (0: lanes 0-15, 1: lanes 16-31), element e in 0..15:
//   vgpr v = e/2;  v<4: K = 8h + 2v + (e&1);  v>=4: K = 16 + 8h + 2(v-4) + (e&1)
__device__ __forceinline__ int wmma_kmap(int e, int h) {
    int v = e >> 1;
    int lo = e & 1;
    return (v < 4) ? (h * 8 + 2 * v + lo) : (16 + h * 8 + 2 * (v - 4) + lo);
}

// Wave-local LDS ordering fence: LDS ops from one wave complete in order, so a
// compiler reorder barrier is sufficient for per-wave staging buffers.
__device__ __forceinline__ void wave_lds_fence() {
    asm volatile("" ::: "memory");
    __builtin_amdgcn_wave_barrier();
    asm volatile("" ::: "memory");
}

// CDNA5 async global->LDS copy (ASYNCcnt-tracked). Low 32 bits of a generic
// LDS pointer are the LDS byte address (ISA 10.2 aperture mapping).
__device__ __forceinline__ void async_copy_b128(void* ldsDst, const void* gSrc) {
    unsigned l = (unsigned)(uintptr_t)ldsDst;
    asm volatile("global_load_async_to_lds_b128 %0, %1, off"
                 :: "v"(l), "v"(gSrc) : "memory");
}
__device__ __forceinline__ void async_wait0() {
    asm volatile("s_wait_asynccnt 0x0" ::: "memory");
}

// ============================================================
// Kernel 1: sf = W_sr(64x128) @ concat(contour, enc_score)  (bf16 WMMA)
// Output layout: sf[pixGlobal][64] bf16 (channel-last)
// ============================================================
__global__ void k_reduce_score(const float* __restrict__ cf,
                               const float* __restrict__ esf,
                               const float* __restrict__ w, // [64][128]
                               __bf16* __restrict__ sf) {
    __shared__ __align__(32) __bf16 Wfrag[4 * 4 * 32 * 16]; // fragment-order, 16 KB
    __shared__ __align__(16) __bf16 Al[8][16][32];          // per-wave A tile, 8 KB
    __shared__ __align__(16) __bf16 St[8][16][64];          // per-wave store repack, 16 KB
    int tid = threadIdx.x;
    for (int i = tid; i < 8192; i += 256) {
        int e = i & 15, lane = (i >> 4) & 31, nt = (i >> 9) & 3, kc = i >> 11;
        int h = lane >> 4;
        int ncol = nt * 16 + (lane & 15);
        int ci = kc * 32 + wmma_kmap(e, h);
        Wfrag[i] = (__bf16)w[ncol * 128 + ci];
    }
    __syncthreads();

    int wave = tid >> 5, lane = tid & 31;
    int m = lane & 15, h = lane >> 4;
    long pixBase = (long)blockIdx.x * 128 + wave * 16;
    int n  = (int)(pixBase >> 16);
    int p0 = (int)(pixBase & (HWC - 1));

    v8f acc0 = {}, acc1 = {}, acc2 = {}, acc3 = {};

    for (int kc = 0; kc < 4; ++kc) {
        const float* src = (kc < 2) ? cf : esf;
        int cbase = (kc & 1) * 32;
        #pragma unroll
        for (int c2 = 0; c2 < 16; ++c2) {
            int ch = c2 * 2 + h;
            Al[wave][m][ch] = (__bf16)src[((long)n * 64 + cbase + ch) * HWC + p0 + m];
        }
        wave_lds_fence();
        v16bf a;
        #pragma unroll
        for (int e = 0; e < 16; ++e) a[e] = Al[wave][m][wmma_kmap(e, h)];
        #pragma unroll
        for (int nt = 0; nt < 4; ++nt) {
            v16bf b = *reinterpret_cast<const v16bf*>(&Wfrag[((kc * 4 + nt) * 32 + lane) * 16]);
            v8f c = (nt == 0) ? acc0 : (nt == 1) ? acc1 : (nt == 2) ? acc2 : acc3;
            c = __builtin_amdgcn_wmma_f32_16x16x32_bf16(false, a, false, b,
                                                        (short)0, c, false, false);
            if (nt == 0) acc0 = c; else if (nt == 1) acc1 = c;
            else if (nt == 2) acc2 = c; else acc3 = c;
        }
        wave_lds_fence();
    }
    #pragma unroll
    for (int nt = 0; nt < 4; ++nt) {
        v8f c = (nt == 0) ? acc0 : (nt == 1) ? acc1 : (nt == 2) ? acc2 : acc3;
        #pragma unroll
        for (int r = 0; r < 8; ++r)
            St[wave][r + 8 * h][nt * 16 + m] = (__bf16)c[r];
    }
    wave_lds_fence();
    int px = lane >> 1, half = lane & 1;
    const uint4* s4 = (const uint4*)&St[wave][px][half * 32];
    uint4* g = (uint4*)(sf + (pixBase + px) * 64 + half * 32);
    #pragma unroll
    for (int j = 0; j < 4; ++j) g[j] = s4[j];
}

// ============================================================
// Kernel 3: rf = W_rr(32x64) @ concat(refinement, enc_ref)   (bf16 WMMA)
// ============================================================
__global__ void k_reduce_ref(const float* __restrict__ rff,
                             const float* __restrict__ erf,
                             const float* __restrict__ w, // [32][64]
                             __bf16* __restrict__ rf) {
    __shared__ __align__(32) __bf16 Wfrag[2 * 2 * 32 * 16]; // 4 KB
    __shared__ __align__(16) __bf16 Al[8][16][32];
    __shared__ __align__(16) __bf16 St[8][16][32];
    int tid = threadIdx.x;
    for (int i = tid; i < 2048; i += 256) {
        int e = i & 15, lane = (i >> 4) & 31, nt = (i >> 9) & 1, kc = i >> 10;
        int h = lane >> 4;
        int ncol = nt * 16 + (lane & 15);
        int ci = kc * 32 + wmma_kmap(e, h);
        Wfrag[i] = (__bf16)w[ncol * 64 + ci];
    }
    __syncthreads();

    int wave = tid >> 5, lane = tid & 31;
    int m = lane & 15, h = lane >> 4;
    long pixBase = (long)blockIdx.x * 128 + wave * 16;
    int n  = (int)(pixBase >> 18);
    int p0 = (int)(pixBase & (HWF - 1));

    v8f acc0 = {}, acc1 = {};
    for (int kc = 0; kc < 2; ++kc) {
        const float* src = (kc == 0) ? rff : erf;
        #pragma unroll
        for (int c2 = 0; c2 < 16; ++c2) {
            int ch = c2 * 2 + h;
            Al[wave][m][ch] = (__bf16)src[((long)n * 32 + ch) * HWF + p0 + m];
        }
        wave_lds_fence();
        v16bf a;
        #pragma unroll
        for (int e = 0; e < 16; ++e) a[e] = Al[wave][m][wmma_kmap(e, h)];
        #pragma unroll
        for (int nt = 0; nt < 2; ++nt) {
            v16bf b = *reinterpret_cast<const v16bf*>(&Wfrag[((kc * 2 + nt) * 32 + lane) * 16]);
            v8f c = (nt == 0) ? acc0 : acc1;
            c = __builtin_amdgcn_wmma_f32_16x16x32_bf16(false, a, false, b,
                                                        (short)0, c, false, false);
            if (nt == 0) acc0 = c; else acc1 = c;
        }
        wave_lds_fence();
    }
    #pragma unroll
    for (int nt = 0; nt < 2; ++nt) {
        v8f c = (nt == 0) ? acc0 : acc1;
        #pragma unroll
        for (int r = 0; r < 8; ++r)
            St[wave][r + 8 * h][nt * 16 + m] = (__bf16)c[r];
    }
    wave_lds_fence();
    int px = lane >> 1, half = lane & 1;
    const uint4* s4 = (const uint4*)&St[wave][px][half * 16];
    uint4* g = (uint4*)(rf + (pixBase + px) * 32 + half * 16);
    #pragma unroll
    for (int j = 0; j < 2; ++j) g[j] = s4[j];
}

// ============================================================
// WMMA 7x7 conv, 2 output channels, via shifted-partial decomposition:
//   z[qx][(oc,kx)] = sum_{ky,c} T[row+ky][qx][c] * w[oc][c][ky][kx]   (WMMA)
//   y[px][oc]      = sum_{kx}   z[px+kx][(oc,kx)]                     (epilogue)
// Block: 128 threads = 4 waves, one output row per wave, 16x4 outputs.
// M tiles cover halo cols 0..31 (22 valid; overread rows of z discarded).
// MODE 0: fg = sigmoid(y1-y0) -> out[n][H*W]
// MODE 1: tanh(y)*3           -> out[n][2][H*W]
// ============================================================
template <int C, int MODE>
__global__ void k_conv7(const __bf16* __restrict__ in,   // [N][H*W][C]
                        const float* __restrict__ w,     // [2][C][7][7]
                        const float* __restrict__ bias,  // [2]
                        float* __restrict__ out,
                        int W, int H) {
    const int NKC = 7 * C / 32;                // k-chunks of 32
    const int FROWS = 9 * 22 + 32;             // flat halo rows + overread pad
    __shared__ __align__(32) __bf16 tileF[FROWS][C];
    __shared__ __align__(32) __bf16 Bf[NKC * 32 * 16];
    __shared__ float zbuf[4][22][14];
    __shared__ float ybuf[4][16][2];

    int n = blockIdx.z;
    int bx = blockIdx.x * 16, by = blockIdx.y * 4;
    int tid = threadIdx.x;
    int wave = tid >> 5, lane = tid & 31;
    int m = lane & 15, h = lane >> 4;

    // Stage B (weights) in per-lane fragment order
    for (int i = tid; i < NKC * 32 * 16; i += 128) {
        int e = i & 15, ln = (i >> 4) & 31, kc = i >> 9;
        int hh = ln >> 4, ncol = ln & 15;
        float v = 0.f;
        if (ncol < 14) {
            int oc = ncol / 7, kx = ncol % 7;
            int k = kc * 32 + wmma_kmap(e, hh);
            int ky = k / C, c = k % C;
            v = w[(oc * C + c) * 49 + ky * 7 + kx];
        }
        Bf[i] = (__bf16)v;
    }
    // Stage halo tile (10 rows x 22 cols) via async global->LDS
    for (int i = tid; i < 10 * 22; i += 128) {
        int ry = i / 22, rx = i % 22;
        int iy = by + ry - 3, ix = bx + rx - 3;
        char* dst = (char*)&tileF[ry * 22 + rx][0];
        if (iy >= 0 && iy < H && ix >= 0 && ix < W) {
            const char* src = (const char*)(in + ((long)n * H * W + (long)iy * W + ix) * C);
            #pragma unroll
            for (int c = 0; c < C * 2 / 16; ++c) async_copy_b128(dst + 16 * c, src + 16 * c);
        } else {
            uint4 z4 = {0u, 0u, 0u, 0u};
            uint4* d4 = (uint4*)dst;
            #pragma unroll
            for (int c = 0; c < C * 2 / 16; ++c) d4[c] = z4;
        }
    }
    async_wait0();
    __syncthreads();

    // z GEMM for output row `wave`
    v8f acc0 = {}, acc1 = {};
    #pragma unroll
    for (int kc = 0; kc < NKC; ++kc) {
        int ky = (C == 64) ? (kc >> 1) : kc;
        int cbase = (C == 64) ? ((kc & 1) * 32) : 0;
        v16bf b = *reinterpret_cast<const v16bf*>(&Bf[(kc * 32 + lane) * 16]);
        #pragma unroll
        for (int t = 0; t < 2; ++t) {
            const __bf16* p = &tileF[(wave + ky) * 22 + 16 * t + m][cbase];
            v16bf a;
            #pragma unroll
            for (int e = 0; e < 16; ++e) a[e] = p[wmma_kmap(e, h)];
            v8f cc = t ? acc1 : acc0;
            cc = __builtin_amdgcn_wmma_f32_16x16x32_bf16(false, a, false, b,
                                                         (short)0, cc, false, false);
            if (t) acc1 = cc; else acc0 = cc;
        }
    }
    // Spill z: C-tile element (t, r) -> qx = 16t + r + 8h, j = lane&15
    #pragma unroll
    for (int t = 0; t < 2; ++t) {
        v8f cc = t ? acc1 : acc0;
        #pragma unroll
        for (int r = 0; r < 8; ++r) {
            int qx = 16 * t + r + 8 * h;
            if (qx < 22 && m < 14) zbuf[wave][qx][m] = cc[r];
        }
    }
    wave_lds_fence();
    // Epilogue: lane -> (px = m, oc = h)
    float y = bias[h];
    #pragma unroll
    for (int kx = 0; kx < 7; ++kx) y += zbuf[wave][m + kx][h * 7 + kx];
    int py = by + wave, px = bx + m;
    if (MODE == 0) {
        ybuf[wave][m][h] = y;
        wave_lds_fence();
        if (h == 0) {
            float a0 = ybuf[wave][m][0], a1 = ybuf[wave][m][1];
            out[(long)n * H * W + (long)py * W + px] = 1.0f / (1.0f + __expf(a0 - a1));
        }
    } else {
        out[((long)n * 2 + h) * (long)H * W + (long)py * W + px] = tanhf(y) * 3.0f;
    }
}

// ============================================================
// Kernel 5: per-batch top-512 (descending score, ties by index)
// ============================================================
__global__ void k_topk(const float* __restrict__ fg,
                       float* __restrict__ scores_out,
                       int* __restrict__ idx_out) {
    __shared__ int hist[4096];
    __shared__ float cval[2048];
    __shared__ int cidx[2048];
    __shared__ int scnt, binB;
    int n = blockIdx.x, tid = threadIdx.x; // 1024 threads
    const float* f = fg + (long)n * HWC;
    for (int i = tid; i < 4096; i += 1024) hist[i] = 0;
    if (tid == 0) scnt = 0;
    __syncthreads();
    for (int i = tid; i < HWC; i += 1024) {
        unsigned bits = __float_as_uint(f[i]);
        atomicAdd(&hist[bits >> 20], 1);
    }
    __syncthreads();
    if (tid == 0) {
        int c = 0, B = 0;
        for (int bb = 4095; bb >= 0; --bb) {
            c += hist[bb];
            if (c >= TOPK) { B = bb; break; }
        }
        binB = B;
    }
    __syncthreads();
    int B = binB;
    for (int i = tid; i < HWC; i += 1024) {
        unsigned bits = __float_as_uint(f[i]);
        if ((int)(bits >> 20) >= B) {
            int s = atomicAdd(&scnt, 1);
            if (s < 2048) { cval[s] = f[i]; cidx[s] = i; }
        }
    }
    __syncthreads();
    int cnt = scnt; if (cnt > 2048) cnt = 2048;
    for (int i = tid; i < 2048; i += 1024)
        if (i >= cnt) { cval[i] = -1.0f; cidx[i] = 0x7fffffff; }
    __syncthreads();
    for (int ksz = 2; ksz <= 2048; ksz <<= 1) {
        for (int j = ksz >> 1; j > 0; j >>= 1) {
            for (int i = tid; i < 2048; i += 1024) {
                int ixj = i ^ j;
                if (ixj > i) {
                    bool up = (i & ksz) == 0;
                    float va = cval[i], vb = cval[ixj];
                    int ia = cidx[i], ib = cidx[ixj];
                    bool aFirst = (va > vb) || (va == vb && ia < ib);
                    if (up ? !aFirst : aFirst) {
                        cval[i] = vb; cval[ixj] = va;
                        cidx[i] = ib; cidx[ixj] = ia;
                    }
                }
            }
            __syncthreads();
        }
    }
    for (int i = tid; i < TOPK; i += 1024) {
        scores_out[n * TOPK + i] = cval[i];
        idx_out[n * TOPK + i] = cidx[i];
    }
}

// ============================================================
// Kernel 6: sparse loc + fourier 7x7 conv at top-k pixels, contour synth
// ============================================================
__global__ void k_heads(const float* __restrict__ cf,
                        const float* __restrict__ wloc, const float* __restrict__ bloc,
                        const float* __restrict__ wfour, const float* __restrict__ bfour,
                        const int* __restrict__ topidx,
                        float* __restrict__ contour) {
    __shared__ float red[64][26];
    __shared__ float coef[26];
    int blk = blockIdx.x;
    int n = blk >> 9, cand = blk & 511;
    int tid = threadIdx.x; // 64: input channel
    int pidx = topidx[n * TOPK + cand];
    int gy = pidx >> 8, gx = pidx & 255;
    float acc[26];
    #pragma unroll
    for (int o = 0; o < 26; ++o) acc[o] = 0.f;
    const float* base = cf + ((long)n * 64 + tid) * HWC;
    for (int ky = 0; ky < 7; ++ky) {
        int iy = gy - 3 + ky;
        if (iy < 0 || iy >= HC) continue;
        for (int kx = 0; kx < 7; ++kx) {
            int ix = gx - 3 + kx;
            if (ix < 0 || ix >= WC) continue;
            float v = base[iy * WC + ix];
            int tap = ky * 7 + kx;
            acc[0] += v * wloc[(0 * 64 + tid) * 49 + tap];
            acc[1] += v * wloc[(1 * 64 + tid) * 49 + tap];
            #pragma unroll
            for (int o = 0; o < 24; ++o)
                acc[2 + o] += v * wfour[((long)o * 64 + tid) * 49 + tap];
        }
    }
    #pragma unroll
    for (int o = 0; o < 26; ++o) red[tid][o] = acc[o];
    __syncthreads();
    for (int s = 32; s > 0; s >>= 1) {
        if (tid < s) {
            #pragma unroll
            for (int o = 0; o < 26; ++o) red[tid][o] += red[tid + s][o];
        }
        __syncthreads();
    }
    if (tid < 26) {
        float b2 = (tid < 2) ? bloc[tid] : bfour[tid - 2];
        coef[tid] = red[0][tid] + b2;
    }
    __syncthreads();
    float cx = ((float)gx + coef[0]) * 2.0f;
    float cy = ((float)gy + coef[1]) * 2.0f;
    float t = (float)tid / (float)SAMPLES;
    float px = 0.f, py = 0.f;
    #pragma unroll
    for (int k = 0; k < ORDER; ++k) {
        float ang = 6.28318530717958647692f * (float)(k + 1) * t;
        float cc = __cosf(ang), ss = __sinf(ang);
        px += coef[2 + k * 4 + 0] * cc + coef[2 + k * 4 + 1] * ss;
        py += coef[2 + k * 4 + 2] * cc + coef[2 + k * 4 + 3] * ss;
    }
    long o = ((long)blk * SAMPLES + tid) * 2;
    contour[o]     = px * 2.0f + cx;
    contour[o + 1] = py * 2.0f + cy;
}

// ============================================================
// Kernel 7: 4 refine iterations + box min/max per candidate
// ============================================================
__global__ void k_refine(const float* __restrict__ refm,
                         float* __restrict__ contour,
                         float* __restrict__ boxes) {
    __shared__ float sx1[64], sy1[64], sx2[64], sy2[64];
    int blk = blockIdx.x;
    int n = blk >> 9;
    int tid = threadIdx.x;
    long o = ((long)blk * SAMPLES + tid) * 2;
    float x = contour[o], y = contour[o + 1];
    const float* r0 = refm + ((long)n * 2 + 0) * HWF;
    const float* r1 = refm + ((long)n * 2 + 1) * HWF;
    #pragma unroll
    for (int it = 0; it < 4; ++it) {
        int xi = (int)fminf(fmaxf(rintf(x), 0.f), (float)(WF - 1));
        int yi = (int)fminf(fmaxf(rintf(y), 0.f), (float)(HF - 1));
        int fl = yi * WF + xi;
        x += r0[fl];
        y += r1[fl];
    }
    contour[o] = x; contour[o + 1] = y;
    sx1[tid] = x; sy1[tid] = y; sx2[tid] = x; sy2[tid] = y;
    __syncthreads();
    for (int s = 32; s > 0; s >>= 1) {
        if (tid < s) {
            sx1[tid] = fminf(sx1[tid], sx1[tid + s]);
            sy1[tid] = fminf(sy1[tid], sy1[tid + s]);
            sx2[tid] = fmaxf(sx2[tid], sx2[tid + s]);
            sy2[tid] = fmaxf(sy2[tid], sy2[tid + s]);
        }
        __syncthreads();
    }
    if (tid == 0) {
        long bo = (long)blk * 4;
        boxes[bo] = sx1[0]; boxes[bo + 1] = sy1[0];
        boxes[bo + 2] = sx2[0]; boxes[bo + 3] = sy2[0];
    }
}

// ============================================================
// Kernel 8: NMS, sequential-i parallel-j
// ============================================================
__global__ void k_nms(const float* __restrict__ boxes,
                      const float* __restrict__ scores,
                      float* __restrict__ keep_out) {
    __shared__ float bx1[512], by1[512], bx2[512], by2[512], ar[512];
    __shared__ int keep[512];
    int n = blockIdx.x, tid = threadIdx.x;
    long bo = (long)(n * TOPK + tid) * 4;
    float x1 = boxes[bo], y1 = boxes[bo + 1], x2 = boxes[bo + 2], y2 = boxes[bo + 3];
    bx1[tid] = x1; by1[tid] = y1; bx2[tid] = x2; by2[tid] = y2;
    ar[tid] = fmaxf(x2 - x1, 0.f) * fmaxf(y2 - y1, 0.f);
    keep[tid] = (scores[n * TOPK + tid] > 0.5f) ? 1 : 0;
    __syncthreads();
    for (int i = 0; i < TOPK; ++i) {
        if (keep[i] && tid > i && keep[tid]) {
            float ix1 = fmaxf(x1, bx1[i]), iy1 = fmaxf(y1, by1[i]);
            float ix2 = fminf(x2, bx2[i]), iy2 = fminf(y2, by2[i]);
            float inter = fmaxf(ix2 - ix1, 0.f) * fmaxf(iy2 - iy1, 0.f);
            float uni = ar[tid] + ar[i] - inter;
            if (inter / fmaxf(uni, 1e-8f) > 0.2f) keep[tid] = 0;
        }
        __syncthreads();
    }
    keep_out[n * TOPK + tid] = (float)keep[tid];
}

// ============================================================
extern "C" void kernel_launch(void* const* d_in, const int* in_sizes, int n_in,
                              void* d_out, int out_size, void* d_ws, size_t ws_size,
                              hipStream_t stream) {
    const float* cf  = (const float*)d_in[0];
    const float* esf = (const float*)d_in[1];
    const float* rff = (const float*)d_in[2];
    const float* erf = (const float*)d_in[3];
    const float* wsr = (const float*)d_in[4];
    const float* wsc = (const float*)d_in[5];
    const float* bsc = (const float*)d_in[6];
    const float* wlo = (const float*)d_in[7];
    const float* blo = (const float*)d_in[8];
    const float* wfo = (const float*)d_in[9];
    const float* bfo = (const float*)d_in[10];
    const float* wrr = (const float*)d_in[11];
    const float* wrf = (const float*)d_in[12];
    const float* brf = (const float*)d_in[13];

    char* ws = (char*)d_ws;
    __bf16* sf  = (__bf16*)(ws);                 // 4*65536*64*2  = 33,554,432 B
    __bf16* rf  = (__bf16*)(ws + 33554432);      // 4*262144*32*2 = 67,108,864 B
    float*  fg  = (float*)(ws + 100663296);      // 4*65536*4     =  1,048,576 B
    float*  rm  = (float*)(ws + 101711872);      // 4*2*262144*4  =  8,388,608 B
    int*    tix = (int*)(ws + 110100480);        // 4*512*4

    float* out     = (float*)d_out;
    float* contour = out;            // 4*512*64*2 = 262144
    float* scores  = out + 262144;   // 4*512      = 2048
    float* keepo   = out + 264192;   // 4*512      = 2048
    float* boxes   = out + 266240;   // 4*512*4    = 8192

    k_reduce_score<<<2048, 256, 0, stream>>>(cf, esf, wsr, sf);
    k_conv7<64, 0><<<dim3(WC / 16, HC / 4, 4), 128, 0, stream>>>(sf, wsc, bsc, fg, WC, HC);
    k_reduce_ref<<<8192, 256, 0, stream>>>(rff, erf, wrr, rf);
    k_conv7<32, 1><<<dim3(WF / 16, HF / 4, 4), 128, 0, stream>>>(rf, wrf, brf, rm, WF, HF);
    k_topk<<<4, 1024, 0, stream>>>(fg, scores, tix);
    k_heads<<<2048, 64, 0, stream>>>(cf, wlo, blo, wfo, bfo, tix, contour);
    k_refine<<<2048, 64, 0, stream>>>(rm, contour, boxes);
    k_nms<<<4, 512, 0, stream>>>(boxes, scores, keepo);
}
